// TransformerEncoderLayer_28587302322304
// MI455X (gfx1250) — matrix operand order repaired and checked by reading, not compiled, not run on hardware
//
#include <hip/hip_runtime.h>
#include <hip/hip_bf16.h>
#include <math.h>

#define B_ 4
#define T_ 1024
#define D_ 256
#define H_ 8
#define DK_ 32
#define DV_ 32
#define C_ 16
#define EPS_SM 1e-5f
#define EPS_LN 1e-5f
#define SLOPE 0.01f

typedef __attribute__((ext_vector_type(16))) _Float16 v16h;
typedef __attribute__((ext_vector_type(8)))  _Float16 v8h;
typedef __attribute__((ext_vector_type(8)))  float    v8f;
typedef __attribute__((ext_vector_type(4)))  int      v4i;

#if defined(__has_builtin)
#if __has_builtin(__builtin_amdgcn_global_load_async_to_lds_b128)
#define HAVE_ASYNC_LDS 1
#endif
#endif
#ifndef HAVE_ASYNC_LDS
#define HAVE_ASYNC_LDS 0
#endif

static __device__ __forceinline__ int lane_id() { return (int)(threadIdx.x & 31u); }

// 16-byte global -> LDS copy; async (ASYNCcnt) when available.
static __device__ __forceinline__ void async_copy16(const _Float16* g, _Float16* l) {
#if HAVE_ASYNC_LDS
  __builtin_amdgcn_global_load_async_to_lds_b128(
      (__attribute__((address_space(1))) v4i*)(void*)g,
      (__attribute__((address_space(3))) v4i*)(void*)l, 0, 0);
#else
  *(float4*)l = *(const float4*)g;
#endif
}

static __device__ __forceinline__ void wait_async() {
#if HAVE_ASYNC_LDS
  asm volatile("s_wait_asynccnt 0" ::: "memory");
#endif
}

// ---- WMMA f16 fragment loaders (CDNA5 ISA §7.12.2 layouts) ----
// A 16x32: lanes 0-15: M=lane, K={0..7,16..23}; lanes 16-31: M=lane-16, K={8..15,24..31}
// Two 16B chunks per lane (works for LDS or global f16, row stride ld halves).
static __device__ __forceinline__ v16h load_a_frag_h(const _Float16* base, int ld) {
  const int lane = lane_id();
  const int m  = lane & 15;
  const int kh = (lane >> 4) ? 8 : 0;
  const _Float16* p = base + (size_t)m * ld + kh;
  v8h lo = *(const v8h*)(p);
  v8h hi = *(const v8h*)(p + 16);
  return __builtin_shufflevector(lo, hi, 0, 1, 2, 3, 4, 5, 6, 7,
                                 8, 9, 10, 11, 12, 13, 14, 15);
}

// B 32x16 with Bfrag[k][n] = base[n*ld + k] (row-major [N,K] source).
// lanes 0-15: N=lane, K={0..7,16..23}; lanes 16-31: N=lane-16, K={8..15,24..31}.
static __device__ __forceinline__ v16h load_b_frag_nt_h(const _Float16* base, int ld) {
  const int lane = lane_id();
  const int n  = lane & 15;
  const int k0 = (lane >> 4) ? 8 : 0;
  const _Float16* p = base + (size_t)n * ld + k0;
  v8h lo = *(const v8h*)(p);
  v8h hi = *(const v8h*)(p + 16);
  return __builtin_shufflevector(lo, hi, 0, 1, 2, 3, 4, 5, 6, 7,
                                 8, 9, 10, 11, 12, 13, 14, 15);
}

static __device__ __forceinline__ v8f wmma16(v16h a, v16h b, v8f c) {
  return __builtin_amdgcn_wmma_f32_16x16x32_f16(false, a, false, b, (short)0, c,
                                                false, false);
}

// ---- f32 -> f16 conversion ----
__global__ __launch_bounds__(256) void cvt_h_kernel(const float* __restrict__ s,
                                                    _Float16* __restrict__ d, int n) {
  const int i = blockIdx.x * 256 + (int)threadIdx.x;
  if (i < n) d[i] = (_Float16)s[i];
}

// ---- LDS-staged NT GEMM: out = mode(A[M,K] * B[N,K]^T [+ bias]) ----
// block: 256 thr, tile 128(M) x 64(N) x 32(K); wave = 32x32 -> 4 WMMA / k-step.
// MODE 0: f16 out; MODE 1: f16 out, V-transposed [B,H,DV,T] layout;
// MODE 2: f32 out + bias;  MODE 3: f32 out + bias + leakyrelu.
template <int MODE>
__global__ __launch_bounds__(256) void gemm_nt16_kernel(
    const _Float16* __restrict__ A, const _Float16* __restrict__ Bm,
    const float* __restrict__ bias, void* __restrict__ Cout, int M, int N, int K) {
  __shared__ __align__(16) _Float16 Atile[128][32];
  __shared__ __align__(16) _Float16 Btile[64][32];
  const int tid = (int)threadIdx.x;
  const int wave = tid >> 5;
  const int lane = tid & 31;
  const int m0 = blockIdx.x * 128;
  const int n0 = blockIdx.y * 64;
  const int wm = (wave >> 1) * 32;
  const int wn = (wave & 1) * 32;

  const int ar = tid >> 2;            // 0..63
  const int ac = (tid & 3) * 8;       // 0,8,16,24 (halves)

  v8f acc[2][2] = {};
  for (int k0 = 0; k0 < K; k0 += 32) {
    __syncthreads();  // previous compute reads done before overwrite
    async_copy16(A + (size_t)(m0 + ar) * K + k0 + ac, &Atile[ar][ac]);
    async_copy16(A + (size_t)(m0 + ar + 64) * K + k0 + ac, &Atile[ar + 64][ac]);
    async_copy16(Bm + (size_t)(n0 + ar) * K + k0 + ac, &Btile[ar][ac]);
    wait_async();
    __syncthreads();
    v16h a0 = load_a_frag_h(&Atile[wm][0], 32);
    v16h a1 = load_a_frag_h(&Atile[wm + 16][0], 32);
    v16h b0 = load_b_frag_nt_h(&Btile[wn][0], 32);
    v16h b1 = load_b_frag_nt_h(&Btile[wn + 16][0], 32);
    acc[0][0] = wmma16(a0, b0, acc[0][0]);
    acc[0][1] = wmma16(a0, b1, acc[0][1]);
    acc[1][0] = wmma16(a1, b0, acc[1][0]);
    acc[1][1] = wmma16(a1, b1, acc[1][1]);
  }

  const int nn = lane & 15;
  const int mhalf = (lane >> 4) ? 8 : 0;
#pragma unroll
  for (int mi = 0; mi < 2; ++mi)
#pragma unroll
    for (int ni = 0; ni < 2; ++ni) {
      const int gn = n0 + wn + ni * 16 + nn;
      const float bv = (MODE >= 2) ? bias[gn] : 0.0f;
#pragma unroll
      for (int r = 0; r < 8; ++r) {
        const int row = m0 + wm + mi * 16 + mhalf + r;
        float v = acc[mi][ni][r] + bv;
        if (MODE == 0) {
          ((_Float16*)Cout)[(size_t)row * N + gn] = (_Float16)v;
        } else if (MODE == 1) {  // V transposed: [B,H,DV,T]
          const int bb = row >> 10, t = row & (T_ - 1);
          const int hh = gn >> 5, dv = gn & 31;
          ((_Float16*)Cout)[((size_t)(bb * H_ + hh) * DV_ + dv) * T_ + t] = (_Float16)v;
        } else {
          if (MODE == 3) v = (v >= 0.0f) ? v : SLOPE * v;
          ((float*)Cout)[(size_t)row * N + gn] = v;
        }
      }
    }
}

// ---- in-place L2 normalize groups of 32 halves (one wave each) ----
__global__ __launch_bounds__(256) void l2norm_h_kernel(_Float16* __restrict__ buf,
                                                       int ngroups) {
  const int g = blockIdx.x * 8 + (int)(threadIdx.x >> 5);
  if (g >= ngroups) return;
  const int lane = lane_id();
  const float v = (float)buf[(size_t)g * 32 + lane];
  float s = v * v;
#pragma unroll
  for (int off = 16; off > 0; off >>= 1) s += __shfl_xor(s, off, 32);
  buf[(size_t)g * 32 + lane] = (_Float16)(v / fmaxf(sqrtf(s), 1e-12f));
}

// ---- fused attention: block stages shared K / V^T tiles (async->LDS),
//      each wave owns 16 t-rows: QK^T WMMA -> pos-bias MLP -> mask -> exp ->
//      streaming no-max softmax -> attn*V WMMA.  f16 in, f16 heads out. ----
__global__ __launch_bounds__(256) void attn_kernel(
    const _Float16* __restrict__ qbuf, const _Float16* __restrict__ kbuf,
    const _Float16* __restrict__ vT, const float* __restrict__ rel,
    const unsigned char* __restrict__ mask,
    const float* __restrict__ pw1, const float* __restrict__ pb1,
    const float* __restrict__ pw2, const float* __restrict__ pb2,
    _Float16* __restrict__ heads) {
  __shared__ __align__(16) _Float16 zbuf[8][16 * 32];  // per-wave z tile
  __shared__ __align__(16) _Float16 ktile[32][32];     // [s_rel][kk]
  __shared__ __align__(16) _Float16 vtile[32][32];     // [dv][s_rel]
  const int tid = (int)threadIdx.x;
  const int wave = tid >> 5;
  const int lane = tid & 31;
  const int b = blockIdx.x >> 3;
  const int h = blockIdx.x & 7;
  const int t0 = (blockIdx.y * 8 + wave) * 16;

  float w1[C_], b1v[C_], w2[C_];
#pragma unroll
  for (int c = 0; c < C_; ++c) {
    w1[c]  = pw1[h * C_ + c];
    b1v[c] = pb1[h * C_ + c];
    w2[c]  = pw2[h * C_ + c];
  }
  const float b2 = pb2[h];

  const _Float16* qh = qbuf + (size_t)(b * T_) * D_ + h * DK_;
  const _Float16* kh = kbuf + (size_t)(b * T_) * D_ + h * DK_;
  const _Float16* vh = vT + (size_t)(b * H_ + h) * DV_ * T_;
  const float* relb = rel + (size_t)b * T_ * T_;
  const unsigned char* mb = mask + (size_t)b * T_ * T_;

  const v16h qfrag = load_a_frag_h(qh + (size_t)t0 * D_, D_);  // K=DK=32

  v8f num0 = {}, num1 = {};
  float rowsum[8];
#pragma unroll
  for (int r = 0; r < 8; ++r) rowsum[r] = 0.0f;

  const int nn = lane & 15;
  const int mh = (lane >> 4) ? 8 : 0;
  _Float16* zslice = &zbuf[wave][0];

  // cooperative staging: 128 chunks K tile + 128 chunks V tile
  const int cr = (tid & 127) >> 2;       // row 0..31
  const int cc = (tid & 3) * 8;          // half-offset 0,8,16,24

  for (int s0 = 0; s0 < T_; s0 += 32) {
    if (tid < 128)
      async_copy16(kh + (size_t)(s0 + cr) * D_ + cc, &ktile[cr][cc]);
    else
      async_copy16(vh + (size_t)cr * T_ + s0 + cc, &vtile[cr][cc]);
    wait_async();
    __syncthreads();

#pragma unroll
    for (int sub = 0; sub < 2; ++sub) {
      const int s = s0 + sub * 16;
      v16h kf = load_b_frag_nt_h(&ktile[sub * 16][0], 32);
      v8f zf = {};
      zf = wmma16(qfrag, kf, zf);
      const int scol = s + nn;
#pragma unroll
      for (int r = 0; r < 8; ++r) {
        const int trow = t0 + mh + r;
        const float d = relb[(size_t)trow * T_ + scol];
        float bias = b2;
#pragma unroll
        for (int c = 0; c < C_; ++c) {
          const float hh2 = fmaf(d, w1[c], b1v[c]);
          bias = fmaf(w2[c], (hh2 >= 0.0f) ? hh2 : SLOPE * hh2, bias);
        }
        const float z =
            mb[(size_t)trow * T_ + scol] ? 0.0f : __expf(zf[r] - bias);
        rowsum[r] += z;
        zslice[(mh + r) * 32 + sub * 16 + nn] = (_Float16)z;
      }
    }
    // wave-private LDS round trip: drain DS stores, reload as A-fragment
    asm volatile("s_wait_dscnt 0" ::: "memory");
    v16h af = load_a_frag_h(zslice, 32);
    v16h v0 = load_b_frag_nt_h(&vtile[0][0], 32);   // Bfrag[s_rel][dv 0..15]
    v16h v1 = load_b_frag_nt_h(&vtile[16][0], 32);  // Bfrag[s_rel][dv 16..31]
    num0 = wmma16(af, v0, num0);
    num1 = wmma16(af, v1, num1);
    __syncthreads();  // all reads done before next iteration's staging
  }

#pragma unroll
  for (int r = 0; r < 8; ++r) {
    float s = rowsum[r];
#pragma unroll
    for (int off = 1; off < 16; off <<= 1) s += __shfl_xor(s, off, 32);
    const float inv = 1.0f / (s + EPS_SM);
    const int trow = t0 + mh + r;
    heads[(size_t)(b * T_ + trow) * D_ + h * DV_ + nn] = (_Float16)(num0[r] * inv);
    heads[(size_t)(b * T_ + trow) * D_ + h * DV_ + 16 + nn] =
        (_Float16)(num1[r] * inv);
  }
}

// ---- out = LayerNorm(a + bsrc)*g + beta ; optional f16 copy of the output ----
__global__ __launch_bounds__(256) void add_ln_kernel(
    const float* __restrict__ a, const float* __restrict__ bsrc,
    const float* __restrict__ g, const float* __restrict__ beta,
    float* __restrict__ outp, _Float16* __restrict__ out16) {
  __shared__ float red[256];
  const int row = blockIdx.x;
  const int t = (int)threadIdx.x;
  const float v = a[(size_t)row * D_ + t] + bsrc[(size_t)row * D_ + t];
  red[t] = v;
  __syncthreads();
  for (int off = 128; off > 0; off >>= 1) {
    if (t < off) red[t] += red[t + off];
    __syncthreads();
  }
  const float mu = red[0] * (1.0f / D_);
  __syncthreads();
  const float dv = v - mu;
  red[t] = dv * dv;
  __syncthreads();
  for (int off = 128; off > 0; off >>= 1) {
    if (t < off) red[t] += red[t + off];
    __syncthreads();
  }
  const float var = red[0] * (1.0f / D_);
  const float o = dv * rsqrtf(var + EPS_LN) * g[t] + beta[t];
  outp[(size_t)row * D_ + t] = o;
  if (out16) out16[(size_t)row * D_ + t] = (_Float16)o;
}

extern "C" void kernel_launch(void* const* d_in, const int* in_sizes, int n_in,
                              void* d_out, int out_size, void* d_ws,
                              size_t ws_size, hipStream_t stream) {
  (void)in_sizes; (void)n_in; (void)out_size; (void)ws_size;
  const float* x    = (const float*)d_in[0];
  const unsigned char* mask = (const unsigned char*)d_in[1];  // jnp.bool_
  const float* rel  = (const float*)d_in[2];
  const float* Wq   = (const float*)d_in[3];
  const float* Wk   = (const float*)d_in[4];
  const float* Wv   = (const float*)d_in[5];
  const float* pw1  = (const float*)d_in[6];
  const float* pb1  = (const float*)d_in[7];
  const float* pw2  = (const float*)d_in[8];
  const float* pb2  = (const float*)d_in[9];
  const float* Wo   = (const float*)d_in[10];
  const float* bo   = (const float*)d_in[11];
  const float* Wf   = (const float*)d_in[12];
  const float* bf   = (const float*)d_in[13];
  const float* g1   = (const float*)d_in[14];
  const float* be1  = (const float*)d_in[15];
  const float* g2   = (const float*)d_in[16];
  const float* be2  = (const float*)d_in[17];
  float* out = (float*)d_out;

  const int M = B_ * T_;              // 4096
  const int MD = M * D_;              // 1048576
  const int WW = D_ * D_;             // 65536

  _Float16* xh     = (_Float16*)d_ws;
  _Float16* wq16   = xh + MD;
  _Float16* wk16   = wq16 + WW;
  _Float16* wv16   = wk16 + WW;
  _Float16* wo16   = wv16 + WW;
  _Float16* wf16   = wo16 + WW;
  _Float16* qh     = wf16 + WW;
  _Float16* kh     = qh + MD;
  _Float16* vT     = kh + MD;
  _Float16* headsh = vT + MD;
  _Float16* zzh    = headsh + MD;
  float* yb = (float*)(zzh + MD);
  float* zz = yb + MD;
  float* zf = zz + MD;

  const dim3 blk(256);

  // f32 -> f16 staging (activations + weights), once
  cvt_h_kernel<<<MD / 256, blk, 0, stream>>>(x, xh, MD);
  cvt_h_kernel<<<WW / 256, blk, 0, stream>>>(Wq, wq16, WW);
  cvt_h_kernel<<<WW / 256, blk, 0, stream>>>(Wk, wk16, WW);
  cvt_h_kernel<<<WW / 256, blk, 0, stream>>>(Wv, wv16, WW);
  cvt_h_kernel<<<WW / 256, blk, 0, stream>>>(Wo, wo16, WW);
  cvt_h_kernel<<<WW / 256, blk, 0, stream>>>(Wf, wf16, WW);

  const dim3 gg(M / 128, D_ / 64);  // 32 x 4
  gemm_nt16_kernel<0><<<gg, blk, 0, stream>>>(xh, wq16, nullptr, qh, M, D_, D_);
  gemm_nt16_kernel<0><<<gg, blk, 0, stream>>>(xh, wk16, nullptr, kh, M, D_, D_);
  gemm_nt16_kernel<1><<<gg, blk, 0, stream>>>(xh, wv16, nullptr, vT, M, D_, D_);

  const int ngrp = M * (D_ / 32);  // 32768
  l2norm_h_kernel<<<ngrp / 8, blk, 0, stream>>>(qh, ngrp);
  l2norm_h_kernel<<<ngrp / 8, blk, 0, stream>>>(kh, ngrp);

  attn_kernel<<<dim3(B_ * H_, T_ / 128), blk, 0, stream>>>(
      qh, kh, vT, rel, mask, pw1, pb1, pw2, pb2, headsh);

  gemm_nt16_kernel<2><<<gg, blk, 0, stream>>>(headsh, wo16, bo, yb, M, D_, D_);
  add_ln_kernel<<<M, blk, 0, stream>>>(x, yb, g1, be1, zz, zzh);
  gemm_nt16_kernel<3><<<gg, blk, 0, stream>>>(zzh, wf16, bf, zf, M, D_, D_);
  add_ln_kernel<<<M, blk, 0, stream>>>(zz, zf, g2, be2, out, nullptr);
}
